// CRF_87806311399531
// MI455X (gfx1250) — matrix operand order repaired and checked by reading, not compiled, not run on hardware
//
#include <hip/hip_runtime.h>
#include <hip/hip_bf16.h>
#include <stdint.h>

// CRF NLL for B=256, S=512, T=128 on gfx1250 (MI455X).
// Forward recursion in exp-space as fp16 WMMA GEMMs, 16 batches/block,
// exp(transitions)^T resident in VGPRs as B-fragments for the whole scan.

#define BB 256
#define SS 512
#define TT 128
#define PF 8   // emission prefetch distance (steps ahead)

typedef __attribute__((ext_vector_type(16))) _Float16 v16h;
typedef __attribute__((ext_vector_type(8)))  float    v8f;

union Frag16 { v16h h; uint4 q[2]; };

// ---------------------------------------------------------------------------
// Kernel 1: ET[n*T + k] = (fp16) exp(transitions[k*T + n])   (transposed exp)
// ---------------------------------------------------------------------------
__global__ __launch_bounds__(256) void crf_prep_exp_trans(
    const float* __restrict__ trans, _Float16* __restrict__ ET) {
  int idx = blockIdx.x * 256 + threadIdx.x;       // 0 .. T*T-1
  int k = idx >> 7;                                // source state
  int n = idx & 127;                               // dest state
  ET[n * TT + k] = (_Float16)__expf(trans[k * TT + n]);
}

// ---------------------------------------------------------------------------
// Kernel 2: gold path score per batch (gathers + block reduction)
// ---------------------------------------------------------------------------
__global__ __launch_bounds__(256) void crf_score(
    const float* __restrict__ emissions, const int* __restrict__ tags,
    const unsigned char* __restrict__ mask, const float* __restrict__ trans,
    const float* __restrict__ start_t, const float* __restrict__ end_t,
    float* __restrict__ score) {
  __shared__ float sred[256];
  __shared__ int   scnt[256];
  const int b = blockIdx.x, tid = threadIdx.x;
  const int* tg = tags + (size_t)b * SS;
  const unsigned char* mk = mask + (size_t)b * SS;
  float acc = 0.f; int cnt = 0;
  for (int s = tid; s < SS; s += 256) {
    int   tc = tg[s];
    float mf = mk[s] ? 1.f : 0.f;
    cnt += mk[s] ? 1 : 0;
    acc += emissions[((size_t)b * SS + s) * TT + tc] * mf;
    if (s > 0) acc += trans[tg[s - 1] * TT + tc] * mf;
  }
  sred[tid] = acc; scnt[tid] = cnt;
  __syncthreads();
  for (int st = 128; st > 0; st >>= 1) {
    if (tid < st) { sred[tid] += sred[tid + st]; scnt[tid] += scnt[tid + st]; }
    __syncthreads();
  }
  if (tid == 0) {
    int last = scnt[0] - 1;
    score[b] = sred[0] + start_t[tg[0]] + end_t[tg[last]];
  }
}

// ---------------------------------------------------------------------------
// Kernel 3: forward scan + partition + final NLL.
// Grid: B/16 blocks, 128 threads (4 waves). Each block handles 16 batches;
// each wave owns 2 N-tiles (32 of the 128 states).
// ---------------------------------------------------------------------------
__global__ __launch_bounds__(128, 1) void crf_forward(
    const float* __restrict__ emissions, const _Float16* __restrict__ ET,
    const float* __restrict__ start_t, const float* __restrict__ end_t,
    const float* __restrict__ score, float* __restrict__ out) {
  __shared__ __align__(16) _Float16 uLDS[2][16 * TT];  // u = exp(fv - m), fp16
  __shared__ float pLDS[2][16][4];                      // per-wave row partials

  const int tid = threadIdx.x;
  const int w    = tid >> 5;        // wave id 0..3
  const int lane = tid & 31;
  const int hid  = (lane >> 4) & 1; // lane half (C/D layout: rows 0-7 vs 8-15)
  const int l16  = lane & 15;
  const int b0   = blockIdx.x * 16;

  // --- B fragments: exp(trans)^T, fp16, resident in VGPRs for the whole scan.
  // Contiguous-K order used here (same as A): lane -> N (=l16 + tile*16), lanes 16-31 hold
  // K+16; vgpr v holds the K pair {2v, 2v+1}.
  Frag16 Bf[4][2];
  const uint32_t* ETw = (const uint32_t*)ET;  // 64 dwords per row (K pairs)
  #pragma unroll
  for (int c = 0; c < 4; ++c)
    #pragma unroll
    for (int i = 0; i < 2; ++i) {
      int n = (2 * w + i) * 16 + l16;
      int base = n * (TT / 2) + c * 16 + hid * 8;
      Bf[c][i].q[0] = *(const uint4*)(ETw + base);
      Bf[c][i].q[1] = *(const uint4*)(ETw + base + 4);
    }

  // Emission base pointers per C/D-layout row (column l16 folded in).
  const float* emb[8];
  #pragma unroll
  for (int r = 0; r < 8; ++r)
    emb[r] = emissions + ((size_t)(b0 + hid * 8 + r) * SS) * TT + l16;

  float stv[2], env[2];
  #pragma unroll
  for (int i = 0; i < 2; ++i) {
    stv[i] = start_t[(2 * w + i) * 16 + l16];
    env[i] = end_t[(2 * w + i) * 16 + l16];
  }

  // fv init at t=0: start + emissions[:,0,:]  (C/D register layout)
  v8f fv[2];
  #pragma unroll
  for (int i = 0; i < 2; ++i)
    #pragma unroll
    for (int r = 0; r < 8; ++r)
      fv[i][r] = stv[i] + __builtin_nontemporal_load(&emb[r][(2 * w + i) * 16]);

  for (int t = 1; t < SS; ++t) {
    const int p = t & 1;
    // 0) prefetch the emission stream PF steps ahead (global_prefetch_b8).
    //    One lane per wave covers this wave's 128B column window per row.
    if (t + PF < SS && l16 == 0) {
      #pragma unroll
      for (int r = 0; r < 8; ++r)
        __builtin_prefetch(&emb[r][(size_t)(t + PF) * TT + 2 * w * 16], 0, 1);
    }
    // 1) row max: local tiles -> 16-lane shfl tree -> cross-wave via LDS
    float pm[8];
    #pragma unroll
    for (int r = 0; r < 8; ++r) pm[r] = fmaxf(fv[0][r], fv[1][r]);
    #pragma unroll
    for (int off = 1; off < 16; off <<= 1)
      #pragma unroll
      for (int r = 0; r < 8; ++r) pm[r] = fmaxf(pm[r], __shfl_xor(pm[r], off, 32));
    if (l16 == 0)
      #pragma unroll
      for (int r = 0; r < 8; ++r) pLDS[p][hid * 8 + r][w] = pm[r];
    __syncthreads();
    float mx[8];
    #pragma unroll
    for (int r = 0; r < 8; ++r) {
      int m = hid * 8 + r;
      mx[r] = fmaxf(fmaxf(pLDS[p][m][0], pLDS[p][m][1]),
                    fmaxf(pLDS[p][m][2], pLDS[p][m][3]));
    }
    // 2) u = exp(fv - mx) -> LDS (fp16, row-major 16 x 128)
    #pragma unroll
    for (int i = 0; i < 2; ++i)
      #pragma unroll
      for (int r = 0; r < 8; ++r)
        uLDS[p][(hid * 8 + r) * TT + (2 * w + i) * 16 + l16] =
            (_Float16)__expf(fv[i][r] - mx[r]);
    __syncthreads();
    // 3) A fragments (contiguous K as in B: lane -> M=l16; lanes 16-31 hold K+16)
    Frag16 Af[4];
    const uint32_t* uw = (const uint32_t*)&uLDS[p][0];
    #pragma unroll
    for (int c = 0; c < 4; ++c) {
      int base = l16 * (TT / 2) + c * 16 + hid * 8;
      Af[c].q[0] = *(const uint4*)(uw + base);
      Af[c].q[1] = *(const uint4*)(uw + base + 4);
    }
    // 4) p = u * exp(trans):  K=128 via 4x v_wmma_f32_16x16x32_f16 per tile
    v8f acc[2] = {};
    #pragma unroll
    for (int c = 0; c < 4; ++c)
      #pragma unroll
      for (int i = 0; i < 2; ++i)
        acc[i] = __builtin_amdgcn_wmma_f32_16x16x32_f16(
            false, Af[c].h, false, Bf[c][i].h, (short)0, acc[i], false, false);
    // 5) fv' = log(p) + mx + emit[t]   (emission loads are single-use -> NT)
    #pragma unroll
    for (int i = 0; i < 2; ++i)
      #pragma unroll
      for (int r = 0; r < 8; ++r)
        fv[i][r] = __logf(acc[i][r]) + mx[r] +
                   __builtin_nontemporal_load(
                       &emb[r][(size_t)t * TT + (2 * w + i) * 16]);
  }

  // partition = logsumexp(fv + end_transitions), then NLL output
  {
    float g[2][8], pm[8];
    #pragma unroll
    for (int i = 0; i < 2; ++i)
      #pragma unroll
      for (int r = 0; r < 8; ++r) g[i][r] = fv[i][r] + env[i];
    #pragma unroll
    for (int r = 0; r < 8; ++r) pm[r] = fmaxf(g[0][r], g[1][r]);
    #pragma unroll
    for (int off = 1; off < 16; off <<= 1)
      #pragma unroll
      for (int r = 0; r < 8; ++r) pm[r] = fmaxf(pm[r], __shfl_xor(pm[r], off, 32));
    if (l16 == 0)
      #pragma unroll
      for (int r = 0; r < 8; ++r) pLDS[0][hid * 8 + r][w] = pm[r];
    __syncthreads();
    float mx[8], ls[8];
    #pragma unroll
    for (int r = 0; r < 8; ++r) {
      int m = hid * 8 + r;
      mx[r] = fmaxf(fmaxf(pLDS[0][m][0], pLDS[0][m][1]),
                    fmaxf(pLDS[0][m][2], pLDS[0][m][3]));
      ls[r] = __expf(g[0][r] - mx[r]) + __expf(g[1][r] - mx[r]);
    }
    #pragma unroll
    for (int off = 1; off < 16; off <<= 1)
      #pragma unroll
      for (int r = 0; r < 8; ++r) ls[r] += __shfl_xor(ls[r], off, 32);
    if (l16 == 0)
      #pragma unroll
      for (int r = 0; r < 8; ++r) pLDS[1][hid * 8 + r][w] = ls[r];
    __syncthreads();
    if (l16 == 0 && w == 0) {
      #pragma unroll
      for (int r = 0; r < 8; ++r) {
        int m = hid * 8 + r;
        float tot = pLDS[1][m][0] + pLDS[1][m][1] + pLDS[1][m][2] + pLDS[1][m][3];
        float part = mx[r] + __logf(tot);
        int b = b0 + m;
        out[b] = -(score[b] - part);
      }
    }
  }
}

// ---------------------------------------------------------------------------
extern "C" void kernel_launch(void* const* d_in, const int* in_sizes, int n_in,
                              void* d_out, int out_size, void* d_ws, size_t ws_size,
                              hipStream_t stream) {
  const float*         emissions = (const float*)d_in[0];         // (B,S,T) f32
  const int*           tags      = (const int*)d_in[1];           // (B,S) i32
  const unsigned char* mask      = (const unsigned char*)d_in[2]; // (B,S) bool
  const float*         trans     = (const float*)d_in[3];         // (T,T) f32
  const float*         start_t   = (const float*)d_in[4];         // (T,) f32
  const float*         end_t     = (const float*)d_in[5];         // (T,) f32
  float*               out       = (float*)d_out;                 // (B,) f32

  _Float16* ET    = (_Float16*)d_ws;                       // T*T fp16 = 32 KB
  float*    score = (float*)((char*)d_ws + 32 * 1024);     // B floats

  crf_prep_exp_trans<<<(TT * TT) / 256, 256, 0, stream>>>(trans, ET);
  crf_score<<<BB, 256, 0, stream>>>(emissions, tags, mask, trans, start_t, end_t, score);
  crf_forward<<<BB / 16, 128, 0, stream>>>(emissions, ET, start_t, end_t, score, out);
}
